// KBGAT_conv_20469814133387
// MI455X (gfx1250) — matrix operand order repaired and checked by reading, compiled, not run on hardware
//
#include <hip/hip_runtime.h>
#include <hip/hip_bf16.h>
#include <math.h>

// ---------------------------------------------------------------------------
// KBGAT conv for MI455X (gfx1250, wave32, WMMA).
// Edge GEMM [E,384]@[384,128] done with v_wmma_f32_16x16x32_bf16 (fp32 accum).
// x / rel / W1 are L2-resident (25.6MB total vs 192MB L2); the kernel is
// L2-gather + atomic bound, so we recompute c in pass3 instead of spilling
// 256MB of c to HBM.
// ---------------------------------------------------------------------------

typedef __attribute__((ext_vector_type(16))) __bf16 v16bf;
typedef __attribute__((ext_vector_type(8)))  float  v8f;

#define DIN   128
#define DOUT  128
#define KTOT  384          // 2*DIN + DREL
#define SLOPE 0.01f
#define TSTR  400          // ushort (bf16) stride of LDS triple tile (16B aligned runs)
#define CSTR  132          // float stride of LDS c tile

// float -> bf16 bits, round-to-nearest-even
__device__ __forceinline__ unsigned short f2bf(float f) {
    unsigned int u = __float_as_uint(f);
    u += 0x7FFFu + ((u >> 16) & 1u);
    return (unsigned short)(u >> 16);
}

// float atomic max via monotone int/uint ordering (init must be -inf)
__device__ __forceinline__ void atomicMaxF(float* addr, float v) {
    if (v >= 0.0f) atomicMax((int*)addr, __float_as_int(v));
    else           atomicMin((unsigned int*)addr, __float_as_uint(v));
}

// ---------------------------------------------------------------------------
// init: zero out[], segmax=-inf, denom=0, and build bf16 W1^T [128][384]
// ---------------------------------------------------------------------------
__global__ void kbgat_init_kernel(const float* __restrict__ W1,
                                  float* __restrict__ out,
                                  float* __restrict__ segmax,
                                  float* __restrict__ denom,
                                  unsigned short* __restrict__ w1t,
                                  int Nn) {
    int stride = gridDim.x * blockDim.x;
    int total  = Nn * DOUT;
    for (int i = blockIdx.x * blockDim.x + threadIdx.x; i < total; i += stride) {
        out[i] = 0.0f;
        if (i < Nn) { segmax[i] = __uint_as_float(0xFF800000u); denom[i] = 0.0f; }
        if (i < DOUT * KTOT) {
            int n = i / KTOT, k = i % KTOT;          // w1t[n][k] = W1[k][n]
            w1t[i] = f2bf(W1[k * DOUT + n]);
        }
    }
}

// ---------------------------------------------------------------------------
// Shared GEMM tile: 16 edges x 128 cols, K=384, 4 waves x 2 wmma tiles each.
// Fills cLds[16][CSTR] with c = triple @ W1 + b1. Ends with __syncthreads().
// ---------------------------------------------------------------------------
__device__ __forceinline__ void edge_gemm_tile(
    const float* __restrict__ x, const float* __restrict__ rel,
    const unsigned short* __restrict__ w1t_u, const float* __restrict__ b1,
    const int* srcS, const int* dstS, const int* typS,
    unsigned short* tripA, float* cLds, int tid)
{
    // gather + convert triple rows into LDS as bf16 (coalesced over k)
    for (int idx = tid; idx < 16 * KTOT; idx += 128) {
        int r = idx / KTOT, k = idx % KTOT;
        float f;
        if (k < DIN)          f = x[(long)srcS[r] * DIN + k];
        else if (k < 2 * DIN) f = x[(long)dstS[r] * DIN + (k - DIN)];
        else                  f = rel[(long)typS[r] * DIN + (k - 2 * DIN)];
        tripA[r * TSTR + k] = f2bf(f);
    }
    __syncthreads();

    int lane = tid & 31, wv = tid >> 5;
    int Mrow = lane & 15, half = lane >> 4;
    int n0 = wv * 32 + (lane & 15);   // global output column of tile 0
    int n1 = n0 + 16;                 // tile 1

    v8f acc0 = {}; v8f acc1 = {};
    const __bf16* w1t = (const __bf16*)w1t_u;

    for (int kk = 0; kk < KTOT / 32; ++kk) {
        int k0 = kk * 32;
        v16bf a, bm0, bm1;
        // A fragment: lane half h holds K in [16h,16h+16)
        const __bf16* pa = (const __bf16*)&tripA[Mrow * TSTR + k0 + half * 16];
        #pragma unroll
        for (int i = 0; i < 8; ++i) { a[i] = pa[i]; a[8 + i] = pa[8 + i]; }
        // B fragment: lanes 0-15 hold K=[k0,k0+16), lanes 16-31 K=[k0+16,k0+32)
        const __bf16* pb0 = w1t + (long)n0 * KTOT + k0 + half * 16;
        const __bf16* pb1 = w1t + (long)n1 * KTOT + k0 + half * 16;
        #pragma unroll
        for (int i = 0; i < 16; ++i) { bm0[i] = pb0[i]; bm1[i] = pb1[i]; }

        acc0 = __builtin_amdgcn_wmma_f32_16x16x32_bf16(false, a, false, bm0,
                                                       (short)0, acc0, false, false);
        acc1 = __builtin_amdgcn_wmma_f32_16x16x32_bf16(false, a, false, bm1,
                                                       (short)0, acc1, false, false);
    }

    // C/D layout: vgpr r, lanes 0-15 -> M=r ; lanes 16-31 -> M=8+r ; N=lane%16
    float bias0 = b1[n0], bias1 = b1[n1];
    #pragma unroll
    for (int r = 0; r < 8; ++r) {
        int row = r + half * 8;
        cLds[row * CSTR + n0] = acc0[r] + bias0;
        cLds[row * CSTR + n1] = acc1[r] + bias1;
    }
    __syncthreads();
}

// ---------------------------------------------------------------------------
// pass1: c -> b = leaky_relu(c.w2) -> bArr, atomic segmax per dst
// ---------------------------------------------------------------------------
__global__ __launch_bounds__(128) void kbgat_pass1_kernel(
    const float* __restrict__ x, const float* __restrict__ rel,
    const unsigned short* __restrict__ w1t, const float* __restrict__ b1,
    const float* __restrict__ w2,
    const int* __restrict__ esrc, const int* __restrict__ edst,
    const int* __restrict__ etyp,
    float* __restrict__ bArr, float* __restrict__ segmax, int E)
{
    __shared__ __align__(16) unsigned short tripA[16 * TSTR];
    __shared__ float cLds[16 * CSTR];
    __shared__ int srcS[16], dstS[16], typS[16];

    int tid = threadIdx.x;
    long eBase = (long)blockIdx.x * 16;
    if (eBase >= E) return;                       // uniform
    int V = min(16, (int)(E - eBase));

    if (tid < 16) {
        int e = (int)eBase + min(tid, V - 1);     // clamp tail (dup rows unused)
        srcS[tid] = esrc[e]; dstS[tid] = edst[e]; typS[tid] = etyp[e];
    }
    __syncthreads();

    edge_gemm_tile(x, rel, w1t, b1, srcS, dstS, typS, tripA, cLds, tid);

    if (tid < V) {
        float s = 0.0f;
        #pragma unroll 4
        for (int n = 0; n < DOUT; ++n) s += cLds[tid * CSTR + n] * w2[n];
        float bb = s > 0.0f ? s : SLOPE * s;
        bArr[eBase + tid] = bb;
        atomicMaxF(&segmax[dstS[tid]], bb);
    }
}

// ---------------------------------------------------------------------------
// pass2: ex = exp(b - segmax[dst]); denom[dst] += ex
// ---------------------------------------------------------------------------
__global__ void kbgat_pass2_kernel(const float* __restrict__ bArr,
                                   const int* __restrict__ edst,
                                   const float* __restrict__ segmax,
                                   float* __restrict__ exArr,
                                   float* __restrict__ denom, int E)
{
    int i = blockIdx.x * blockDim.x + threadIdx.x;
    if (i >= E) return;
    int d = edst[i];
    float ex = __expf(bArr[i] - segmax[d]);
    exArr[i] = ex;
    atomicAdd(&denom[d], ex);
}

// ---------------------------------------------------------------------------
// pass3: recompute c (WMMA), alpha = ex/denom[dst], out[dst] += alpha*c
// ---------------------------------------------------------------------------
__global__ __launch_bounds__(128) void kbgat_pass3_kernel(
    const float* __restrict__ x, const float* __restrict__ rel,
    const unsigned short* __restrict__ w1t, const float* __restrict__ b1,
    const int* __restrict__ esrc, const int* __restrict__ edst,
    const int* __restrict__ etyp,
    const float* __restrict__ exArr, const float* __restrict__ denom,
    float* __restrict__ out, int E)
{
    __shared__ __align__(16) unsigned short tripA[16 * TSTR];
    __shared__ float cLds[16 * CSTR];
    __shared__ int srcS[16], dstS[16], typS[16];
    __shared__ float alphaS[16];

    int tid = threadIdx.x;
    long eBase = (long)blockIdx.x * 16;
    if (eBase >= E) return;                       // uniform
    int V = min(16, (int)(E - eBase));

    if (tid < 16) {
        int e = (int)eBase + min(tid, V - 1);
        srcS[tid] = esrc[e]; dstS[tid] = edst[e]; typS[tid] = etyp[e];
    }
    __syncthreads();

    edge_gemm_tile(x, rel, w1t, b1, srcS, dstS, typS, tripA, cLds, tid);

    if (tid < 16) {
        float a_ = 0.0f;
        if (tid < V) {
            float d = denom[dstS[tid]];
            a_ = d > 0.0f ? exArr[eBase + tid] / d : 0.0f;
        }
        alphaS[tid] = a_;
    }
    __syncthreads();

    // column-coalesced scatter: 128 lanes = 128 columns, loop over edges
    for (int e = 0; e < V; ++e) {
        float v = alphaS[e] * cLds[e * CSTR + tid];
        atomicAdd(&out[(long)dstS[e] * DOUT + tid], v);
    }
}

// ---------------------------------------------------------------------------
// final leaky_relu on out
// ---------------------------------------------------------------------------
__global__ void kbgat_final_kernel(float* __restrict__ out, int total) {
    int stride = gridDim.x * blockDim.x;
    for (int i = blockIdx.x * blockDim.x + threadIdx.x; i < total; i += stride) {
        float v = out[i];
        out[i] = v > 0.0f ? v : SLOPE * v;
    }
}

// ---------------------------------------------------------------------------
extern "C" void kernel_launch(void* const* d_in, const int* in_sizes, int n_in,
                              void* d_out, int out_size, void* d_ws, size_t ws_size,
                              hipStream_t stream) {
    (void)n_in; (void)out_size; (void)ws_size;
    const float* x    = (const float*)d_in[0];
    const float* rel  = (const float*)d_in[1];
    const float* W1   = (const float*)d_in[2];
    const float* b1   = (const float*)d_in[3];
    const float* w2   = (const float*)d_in[4];
    const int*   eidx = (const int*)d_in[5];
    const int*   etyp = (const int*)d_in[6];

    const int E  = in_sizes[6];          // edge_type count
    const int Nn = in_sizes[0] / DIN;    // num nodes
    const int* esrc = eidx;
    const int* edst = eidx + E;

    // workspace layout (256B-aligned chunks)
    char* ws = (char*)d_ws;
    size_t o = 0;
    unsigned short* w1t = (unsigned short*)(ws + o);
    o += (size_t)DOUT * KTOT * sizeof(unsigned short); o = (o + 255) & ~(size_t)255;
    float* segmax = (float*)(ws + o); o += (size_t)Nn * 4; o = (o + 255) & ~(size_t)255;
    float* denom  = (float*)(ws + o); o += (size_t)Nn * 4; o = (o + 255) & ~(size_t)255;
    float* bArr   = (float*)(ws + o); o += (size_t)E * 4;  o = (o + 255) & ~(size_t)255;
    float* exArr  = (float*)(ws + o);

    float* out = (float*)d_out;
    const int total = Nn * DOUT;

    kbgat_init_kernel<<<(total + 255) / 256, 256, 0, stream>>>(
        W1, out, segmax, denom, w1t, Nn);

    const int nblk = (E + 15) / 16;
    kbgat_pass1_kernel<<<nblk, 128, 0, stream>>>(
        x, rel, w1t, b1, w2, esrc, edst, etyp, bArr, segmax, E);

    kbgat_pass2_kernel<<<(E + 255) / 256, 256, 0, stream>>>(
        bArr, edst, segmax, exArr, denom, E);

    kbgat_pass3_kernel<<<nblk, 128, 0, stream>>>(
        x, rel, w1t, b1, esrc, edst, etyp, exArr, denom, out, E);

    kbgat_final_kernel<<<(total + 255) / 256, 256, 0, stream>>>(out, total);
}
